// universalNet_73598559584363
// MI455X (gfx1250) — matrix operand order repaired and checked by reading, not compiled, not run on hardware
//
#include <hip/hip_runtime.h>

typedef _Float16 f16;
typedef __attribute__((ext_vector_type(16))) _Float16 v16h;
typedef __attribute__((ext_vector_type(8)))  _Float16 v8h;
typedef __attribute__((ext_vector_type(4)))  _Float16 v4h;
typedef __attribute__((ext_vector_type(8)))  float    v8f;

// Network shape (fixed by reference): B x 128 -> E=128, I=64, OUT=128, T steps.
// Weight fragment blob in d_ws: 96 fragments x 512 halves, each fragment stored
// as two 256-half planes: plane0 = halves 0..7 of lanes 0..31, plane1 = halves 8..15.
#define FR_IN2E  0    // 8 ntiles x 4 ktiles (N=128, K=128)  -> from L2
#define FR_I2E   32   // 8 x 2              (N=128, K=64)    -> cached in LDS
#define FR_E2I   48   // 4 x 4              (N=64,  K=128)   -> cached in LDS
#define FR_E2OUT 64   // 8 x 4              (N=128, K=128)   -> from L2
#define FR_TOTAL 96

#define ESTR 136   // padded row stride (halves) of 16x128 tile: 68 dwords -> conflict-free
#define ISTR 72    // padded row stride (halves) of 16x64  tile: 36 dwords -> conflict-free
#define WAVE_SCR (16*ESTR + 16*ISTR)   // 3328 halves per wave

// ---------------- weight prep: fp32 [out,in] -> f16 plane-layout fragment blob
// B fragment (32x16, f16): lane L -> n = L%16 ; half h<8 -> k = kt*32 + (L>=16?8:0) + h, h>=8 -> k = kt*32 + 16 + (L>=16?8:0) + (h-8)
__global__ void prep_weights(const float* __restrict__ Win2E,
                             const float* __restrict__ WI2E,
                             const float* __restrict__ WE2I,
                             const float* __restrict__ WE2Out,
                             f16* __restrict__ blob) {
  int f = blockIdx.x;           // fragment id 0..95
  int lane = threadIdx.x;       // 0..31
  const float* W; int K, nt, kt;
  if (f < 32)      { int r = f;      W = Win2E;  K = 128; nt = r >> 2; kt = r & 3; }
  else if (f < 48) { int r = f - 32; W = WI2E;   K = 64;  nt = r >> 1; kt = r & 1; }
  else if (f < 64) { int r = f - 48; W = WE2I;   K = 128; nt = r >> 2; kt = r & 3; }
  else             { int r = f - 64; W = WE2Out; K = 128; nt = r >> 2; kt = r & 3; }
  int n     = nt * 16 + (lane & 15);
  int kbase = kt * 32 + ((lane >> 4) << 3);
  f16* fr = blob + f * 512;
#pragma unroll
  for (int h = 0; h < 8; ++h)                       // plane 0: halves 0..7
    fr[lane * 8 + h]       = (f16)W[n * K + kbase + h];
#pragma unroll
  for (int h = 0; h < 8; ++h)                       // plane 1: halves 8..15
    fr[256 + lane * 8 + h] = (f16)W[n * K + kbase + 16 + h];
}

// ---------------- fragment loaders ------------------------------------------
// A fragment (16x32, f16) from row-major f16 tile with padded stride:
// lane L -> m = L%16, delta = 8*(L>=16); halves 0..7 = k in [kt*32+d, +7],
// halves 8..15 = k in [kt*32+16+d, +7]  -> two ds_load_b128, conflict-free.
__device__ __forceinline__ v16h ldA(const f16* base, int stride, int kt, int lane) {
  int m   = lane & 15;
  int dlt = (lane >> 4) << 3;
  const f16* p = base + m * stride + kt * 32 + dlt;
  v16h a;
  *(v8h*)&a         = *(const v8h*)p;
  *(((v8h*)&a) + 1) = *(const v8h*)(p + 16);
  return a;
}

// B fragment from plane-layout storage (global blob or LDS copy): lane stride 16B.
__device__ __forceinline__ v16h ldB(const f16* store, int frag, int lane) {
  const f16* p = store + frag * 512 + lane * 8;
  v16h b;
  *(v8h*)&b         = *(const v8h*)p;
  *(((v8h*)&b) + 1) = *(const v8h*)(p + 256);
  return b;
}

#define WMMA(a, b, c) \
  __builtin_amdgcn_wmma_f32_16x16x32_f16(false, (a), false, (b), (short)0, (c), false, false)

// ---------------- main persistent-tile RNN kernel ---------------------------
__global__ __launch_bounds__(128) void rnn_kernel(const float* __restrict__ x,
                                                  const f16*   __restrict__ blob,
                                                  const int*   __restrict__ Tptr,
                                                  float*       __restrict__ out,
                                                  int numTiles) {
  __shared__ __align__(16) f16 lw[32 * 512];        // hot weights: W_I2E + W_E2I (32 KB)
  __shared__ __align__(16) f16 scratch[4 * WAVE_SCR];
  const int lane = threadIdx.x & 31;
  const int wave = threadIdx.x >> 5;

  // ---- stage hot-loop weight fragments (frags 32..63) into LDS, once per WG ----
  {
    const v8h* src = (const v8h*)(blob + FR_I2E * 512);
    v8h* dst = (v8h*)lw;
#pragma unroll
    for (int i = 0; i < 16; ++i)
      dst[i * 128 + threadIdx.x] = src[i * 128 + threadIdx.x];
  }
  __syncthreads();

  const int tile = blockIdx.x * 4 + wave;
  if (tile >= numTiles) return;                     // wave-uniform: EXEC stays all-1s

  f16* eBuf = scratch + wave * WAVE_SCR;            // [16][ESTR]  (x, then actE)
  f16* iBuf = eBuf + 16 * ESTR;                     // [16][ISTR]

  // ---- stage x tile to LDS as f16 (coalesced float4 loads) ----
  const float4* xs = (const float4*)(x + (size_t)tile * (16 * 128));
#pragma unroll
  for (int i = 0; i < 16; ++i) {
    int idx = i * 32 + lane;                        // flat float4 index in 16x128 tile
    float4 v = xs[idx];
    int row = idx >> 5, c4 = idx & 31;              // 32 float4 per row
    v4h h = {(f16)v.x, (f16)v.y, (f16)v.z, (f16)v.w};
    *(v4h*)(eBuf + row * ESTR + c4 * 4) = h;
  }

  // ---- ff = x @ W_in2E^T  (kept resident in 8 x v8f) ----
  v16h aX[4];
#pragma unroll
  for (int kt = 0; kt < 4; ++kt) aX[kt] = ldA(eBuf, ESTR, kt, lane);

  v8f ff[8];
#pragma unroll
  for (int nt = 0; nt < 8; ++nt) {
    v8f acc = {};
#pragma unroll
    for (int kt = 0; kt < 4; ++kt)
      acc = WMMA(aX[kt], ldB(blob, FR_IN2E + nt * 4 + kt, lane), acc);
    ff[nt] = acc;
  }

  const int mrow = (lane >> 4) << 3;  // 0 or 8 (C/D layout: row = vgpr + 8*hi)
  const int ncol = lane & 15;
  const int T = Tptr[0];

  v16h aI0 = {}, aI1 = {};            // actI fragments (actI0 = 0)
  v16h aE[4] = {};

  for (int t = 0; t < T; ++t) {
    // actE = relu(ff + actI @ W_I2E^T) -> f16 into eBuf   (B from LDS)
#pragma unroll
    for (int nt = 0; nt < 8; ++nt) {
      v8f d = ff[nt];
      d = WMMA(aI0, ldB(lw, nt * 2 + 0, lane), d);
      d = WMMA(aI1, ldB(lw, nt * 2 + 1, lane), d);
#pragma unroll
      for (int v = 0; v < 8; ++v) {
        float r = d[v] > 0.f ? d[v] : 0.f;
        eBuf[(mrow + v) * ESTR + nt * 16 + ncol] = (f16)r;
      }
    }
    // reload actE as A fragments (used by E2I and, on last step, E2Out)
#pragma unroll
    for (int kt = 0; kt < 4; ++kt) aE[kt] = ldA(eBuf, ESTR, kt, lane);

    if (t + 1 < T) {
      // actI = relu(actE @ W_E2I^T) -> f16 into iBuf -> A fragments  (B from LDS)
#pragma unroll
      for (int nt = 0; nt < 4; ++nt) {
        v8f d = {};
#pragma unroll
        for (int kt = 0; kt < 4; ++kt)
          d = WMMA(aE[kt], ldB(lw, 16 + nt * 4 + kt, lane), d);
#pragma unroll
        for (int v = 0; v < 8; ++v) {
          float r = d[v] > 0.f ? d[v] : 0.f;
          iBuf[(mrow + v) * ISTR + nt * 16 + ncol] = (f16)r;
        }
      }
      aI0 = ldA(iBuf, ISTR, 0, lane);
      aI1 = ldA(iBuf, ISTR, 1, lane);
    }
  }

  // ---- out = relu(actE_last @ W_E2Out^T), fp32 store ----
  float* op = out + (size_t)tile * (16 * 128);
#pragma unroll
  for (int nt = 0; nt < 8; ++nt) {
    v8f d = {};
#pragma unroll
    for (int kt = 0; kt < 4; ++kt)
      d = WMMA(aE[kt], ldB(blob, FR_E2OUT + nt * 4 + kt, lane), d);
#pragma unroll
    for (int v = 0; v < 8; ++v) {
      float r = d[v] > 0.f ? d[v] : 0.f;
      op[(mrow + v) * 128 + nt * 16 + ncol] = r;
    }
  }
}

// ---------------- host launcher ---------------------------------------------
extern "C" void kernel_launch(void* const* d_in, const int* in_sizes, int n_in,
                              void* d_out, int out_size, void* d_ws, size_t ws_size,
                              hipStream_t stream) {
  const float* x      = (const float*)d_in[0];
  const float* Win2E  = (const float*)d_in[1];
  const float* WI2E   = (const float*)d_in[2];
  const float* WE2I   = (const float*)d_in[3];
  const float* WE2Out = (const float*)d_in[4];
  const int*   Tptr   = (const int*)d_in[5];
  float* out = (float*)d_out;
  f16* blob = (f16*)d_ws;                 // needs 96*512*2 = 96 KB of scratch

  int B = in_sizes[0] / 128;
  int numTiles = B / 16;                  // B = 131072 -> 8192 tiles

  prep_weights<<<FR_TOTAL, 32, 0, stream>>>(Win2E, WI2E, WE2I, WE2Out, blob);
  rnn_kernel<<<(numTiles + 3) / 4, 128, 0, stream>>>(x, blob, Tptr, out, numTiles);
}
